// GatedCrossModalAttention_20057497272871
// MI455X (gfx1250) — compile-verified, matrix operand order checked by reading, not run on hardware
//
#include <hip/hip_runtime.h>
#include <hip/hip_bf16.h>

#define DD    256
#define NMOD  5
#define MT    64          // batch rows per workgroup
#define NTHR  512         // 16 wave32
#define LNEPS 1e-5f

typedef __attribute__((ext_vector_type(16))) __bf16 v16bf;
typedef __attribute__((ext_vector_type(8)))  float  v8f;

union BFrag {
    v16bf v;
    struct { uint4 lo; uint4 hi; } q;
};

static __device__ __forceinline__ unsigned short f2bf(float f) {
    unsigned u = __float_as_uint(f);
    unsigned r = u + 0x7FFFu + ((u >> 16) & 1u);   // RNE
    return (unsigned short)(r >> 16);
}

// sum across the 16-lane half-wave group that owns one batch row (wave32)
static __device__ __forceinline__ float rsum16(float v) {
    v += __shfl_xor(v, 1, 16);
    v += __shfl_xor(v, 2, 16);
    v += __shfl_xor(v, 4, 16);
    v += __shfl_xor(v, 8, 16);
    return v;
}

// ---------------------------------------------------------------------------
// Kernel 1: W_eff[n] = Wo_n @ Wv_n (bf16), b_eff[n] = Wo_n @ bv_n + bo_n (f32)
// n=0 self, n=1..4 cross[n-1]. 80 blocks x 256 threads.
// ---------------------------------------------------------------------------
__global__ __launch_bounds__(256)
void fuse_weights_kernel(const float* __restrict__ self_Wi, const float* __restrict__ self_bi,
                         const float* __restrict__ self_Wo, const float* __restrict__ self_bo,
                         const float* __restrict__ cross_Wi, const float* __restrict__ cross_bi,
                         const float* __restrict__ cross_Wo, const float* __restrict__ cross_bo,
                         unsigned short* __restrict__ W_eff, float* __restrict__ b_eff)
{
    __shared__ float wo[16 * DD];                  // 16 KB: 16 rows of Wo
    const int tid = threadIdx.x;
    const int n   = blockIdx.x >> 4;               // branch 0..4
    const int o0  = (blockIdx.x & 15) * 16;        // output-row tile

    const float *Wo, *Wv, *bv, *bo;
    if (n == 0) {
        Wo = self_Wo;
        Wv = self_Wi + (size_t)2 * DD * DD;        // value block of packed Wi
        bv = self_bi + 2 * DD;
        bo = self_bo;
    } else {
        const int c = n - 1;
        Wo = cross_Wo + (size_t)c * DD * DD;
        Wv = cross_Wi + (size_t)c * 3 * DD * DD + (size_t)2 * DD * DD;
        bv = cross_bi + (size_t)c * 3 * DD + 2 * DD;
        bo = cross_bo + (size_t)c * DD;
    }

    #pragma unroll
    for (int r = 0; r < 16; ++r)
        wo[r * DD + tid] = Wo[(size_t)(o0 + r) * DD + tid];
    __syncthreads();

    float acc[16];
    #pragma unroll
    for (int r = 0; r < 16; ++r) acc[r] = 0.f;

    for (int j = 0; j < DD; ++j) {
        const float wvk = Wv[(size_t)j * DD + tid];   // coalesced, L2-hot
        #pragma unroll
        for (int r = 0; r < 16; ++r)
            acc[r] = fmaf(wo[r * DD + j], wvk, acc[r]);
    }

    #pragma unroll
    for (int r = 0; r < 16; ++r)
        W_eff[((size_t)n * DD + o0 + r) * DD + tid] = f2bf(acc[r]);

    if (tid < 16) {
        float bias = bo[o0 + tid];
        for (int j = 0; j < DD; ++j)
            bias = fmaf(wo[tid * DD + j], bv[j], bias);
        b_eff[n * DD + o0 + tid] = bias;
    }
}

// ---------------------------------------------------------------------------
// Kernel 2: main fused pass. B/64 workgroups x 512 threads (16 wave32).
// Large-LDS design: ~228 KB/WG (CDNA5 allows 320 KB) so each weight B-frag
// read from L2 is reused across 4 M-tiles.
// ---------------------------------------------------------------------------
__global__ __launch_bounds__(NTHR)
void gcma_main_kernel(const float* __restrict__ x,
                      const float* __restrict__ cross_ln_g, const float* __restrict__ cross_ln_b,
                      const float* __restrict__ gate_ln_g,  const float* __restrict__ gate_ln_b,
                      const float* __restrict__ gate_W,     const float* __restrict__ gate_b,
                      const float* __restrict__ final_ln_g, const float* __restrict__ final_ln_b,
                      const unsigned short* __restrict__ W_eff, const float* __restrict__ b_eff,
                      const int* __restrict__ modal_idx_p,  float* __restrict__ out)
{
    __shared__ float          qf[MT][DD];          // query rows f32     (64 KB)
    __shared__ unsigned short At[MT][DD + 8];      // bf16 A tile, padded (33 KB)
    __shared__ float          Ct[MT][DD + 4];      // GEMM result tile    (65 KB)
    __shared__ float          comb[MT][DD];        // gated accumulator   (64 KB)
    __shared__ float          gates_s[MT][8];      // per-row gates       (2 KB)

    const int tid  = threadIdx.x;
    const int wave = tid >> 5;                     // 0..15 -> N-tile
    const int lane = tid & 31;
    const int Mr   = lane & 15;                    // WMMA M (A) / N (B) index
    const int half = lane >> 4;                    // WMMA K-half select
    const int row16 = tid >> 4;                    // per-row phase: 0..31
    const int l16  = tid & 15;
    const int c0   = l16 * 16;                     // 16 columns per thread
    const int midx = *modal_idx_p;
    const long b0  = (long)blockIdx.x * MT;

    // ---- load query tile: f32 -> qf, bf16 -> At (A-operand for branch 0) ----
    #pragma unroll
    for (int rg = 0; rg < 2; ++rg) {
        const int row = rg * 32 + row16;
        const float* src = x + ((b0 + row) * NMOD + midx) * DD + c0;
        #pragma unroll
        for (int j = 0; j < 16; j += 4) {
            float4 v = *(const float4*)(src + j);
            *(float4*)&qf[row][c0 + j] = v;
            At[row][c0 + j + 0] = f2bf(v.x);
            At[row][c0 + j + 1] = f2bf(v.y);
            At[row][c0 + j + 2] = f2bf(v.z);
            At[row][c0 + j + 3] = f2bf(v.w);
        }
    }
    __syncthreads();

    // ---- gates = softmax(LN(q) @ gate_W^T + gate_b) over NMOD, per row ----
    #pragma unroll
    for (int rg = 0; rg < 2; ++rg) {
        const int row = rg * 32 + row16;
        float qv[16];
        #pragma unroll
        for (int j = 0; j < 16; ++j) qv[j] = qf[row][c0 + j];

        float s1 = 0.f, s2 = 0.f;
        #pragma unroll
        for (int j = 0; j < 16; ++j) { s1 += qv[j]; s2 = fmaf(qv[j], qv[j], s2); }
        s1 = rsum16(s1); s2 = rsum16(s2);
        const float mean = s1 * (1.f / DD);
        const float var  = s2 * (1.f / DD) - mean * mean;
        const float rstd = rsqrtf(var + LNEPS);

        float lq[16];
        #pragma unroll
        for (int j = 0; j < 16; ++j)
            lq[j] = fmaf((qv[j] - mean) * rstd, gate_ln_g[c0 + j], gate_ln_b[c0 + j]);

        float gl[NMOD];
        float mx = -3.4e38f;
        #pragma unroll
        for (int m = 0; m < NMOD; ++m) {
            float p = 0.f;
            #pragma unroll
            for (int j = 0; j < 16; ++j)
                p = fmaf(lq[j], gate_W[m * DD + c0 + j], p);
            gl[m] = rsum16(p) + gate_b[m];
            mx = fmaxf(mx, gl[m]);
        }
        float se = 0.f;
        #pragma unroll
        for (int m = 0; m < NMOD; ++m) { gl[m] = __expf(gl[m] - mx); se += gl[m]; }
        const float inv = 1.f / se;
        #pragma unroll
        for (int m = 0; m < NMOD; ++m) gates_s[row][m] = gl[m] * inv;  // 16 lanes write same value
    }
    __syncthreads();

    #pragma unroll
    for (int n = 0; n < NMOD; ++n) {
        if (n > 0) {
            __syncthreads();                        // iter n-1 consumers of At/Ct done
            const int c = n - 1;
            const int m = c + (c >= midx ? 1 : 0);  // other_idx[c]
            #pragma unroll
            for (int rg = 0; rg < 2; ++rg) {
                const int row = rg * 32 + row16;
                const float* src = x + ((b0 + row) * NMOD + m) * DD + c0;
                #pragma unroll
                for (int j = 0; j < 16; j += 4) {
                    float4 v = *(const float4*)(src + j);
                    At[row][c0 + j + 0] = f2bf(v.x);
                    At[row][c0 + j + 1] = f2bf(v.y);
                    At[row][c0 + j + 2] = f2bf(v.z);
                    At[row][c0 + j + 3] = f2bf(v.w);
                }
            }
            __syncthreads();
        }

        // ---- Ct[64x256] = A[64x256] @ W_eff[n]^T via bf16 WMMA ----
        // wave -> one N-tile; B-frag loaded once per k-step, reused by 4 M-tiles
        const unsigned short* Wn = W_eff + (size_t)n * DD * DD;
        v8f acc[4];
        #pragma unroll
        for (int mt = 0; mt < 4; ++mt) acc[mt] = (v8f){};
        #pragma unroll
        for (int kk = 0; kk < 8; ++kk) {
            const int kb = kk * 32 + half * 8;      // ISA 16-bit frag layout
            BFrag fb;
            const unsigned short* w0 = Wn + (size_t)(wave * 16 + Mr) * DD + kb;
            fb.q.lo = *(const uint4*)(w0);          // K = kb .. kb+7
            fb.q.hi = *(const uint4*)(w0 + 16);     // K = kb+16 .. kb+23
            #pragma unroll
            for (int mt = 0; mt < 4; ++mt) {
                BFrag a;
                a.q.lo = *(const uint4*)&At[mt * 16 + Mr][kb];
                a.q.hi = *(const uint4*)&At[mt * 16 + Mr][kb + 16];
                acc[mt] = __builtin_amdgcn_wmma_f32_16x16x32_bf16(false, a.v, false, fb.v,
                                                                  (short)0, acc[mt], false, false);
            }
        }
        #pragma unroll
        for (int mt = 0; mt < 4; ++mt)              // C/D layout: VGPR r -> M = r + 8*half
            #pragma unroll
            for (int r = 0; r < 8; ++r)
                Ct[mt * 16 + half * 8 + r][wave * 16 + Mr] = acc[mt][r];
        __syncthreads();

        // ---- per-row combine: comb = sum_n gate_n * branch_n ----
        const float* be = b_eff + n * DD;
        #pragma unroll
        for (int rg = 0; rg < 2; ++rg) {
            const int row = rg * 32 + row16;
            const float g = gates_s[row][n];
            if (n == 0) {
                // self branch: no residual/LN; first branch WRITES comb
                #pragma unroll
                for (int j = 0; j < 16; ++j)
                    comb[row][c0 + j] = g * (Ct[row][c0 + j] + be[c0 + j]);
            } else {
                const int cc = n - 1;
                float t[16]; float u1 = 0.f, u2 = 0.f;
                #pragma unroll
                for (int j = 0; j < 16; ++j) {
                    t[j] = qf[row][c0 + j] + Ct[row][c0 + j] + be[c0 + j];
                    u1 += t[j]; u2 = fmaf(t[j], t[j], u2);
                }
                u1 = rsum16(u1); u2 = rsum16(u2);
                const float mn = u1 * (1.f / DD);
                const float vv = u2 * (1.f / DD) - mn * mn;
                const float rs = rsqrtf(vv + LNEPS);
                const float* lg = cross_ln_g + cc * DD;
                const float* lb = cross_ln_b + cc * DD;
                #pragma unroll
                for (int j = 0; j < 16; ++j)
                    comb[row][c0 + j] = fmaf(g, fmaf((t[j] - mn) * rs, lg[c0 + j], lb[c0 + j]),
                                             comb[row][c0 + j]);
            }
        }
    }
    __syncthreads();

    // ---- final LayerNorm(q + combined) ----
    #pragma unroll
    for (int rg = 0; rg < 2; ++rg) {
        const int row = rg * 32 + row16;
        float y[16]; float u1 = 0.f, u2 = 0.f;
        #pragma unroll
        for (int j = 0; j < 16; ++j) {
            y[j] = qf[row][c0 + j] + comb[row][c0 + j];
            u1 += y[j]; u2 = fmaf(y[j], y[j], u2);
        }
        u1 = rsum16(u1); u2 = rsum16(u2);
        const float mn = u1 * (1.f / DD);
        const float vv = u2 * (1.f / DD) - mn * mn;
        const float rs = rsqrtf(vv + LNEPS);
        float* dst = out + (b0 + row) * DD + c0;
        #pragma unroll
        for (int j = 0; j < 16; j += 4) {
            float4 v;
            v.x = fmaf((y[j + 0] - mn) * rs, final_ln_g[c0 + j + 0], final_ln_b[c0 + j + 0]);
            v.y = fmaf((y[j + 1] - mn) * rs, final_ln_g[c0 + j + 1], final_ln_b[c0 + j + 1]);
            v.z = fmaf((y[j + 2] - mn) * rs, final_ln_g[c0 + j + 2], final_ln_b[c0 + j + 2]);
            v.w = fmaf((y[j + 3] - mn) * rs, final_ln_g[c0 + j + 3], final_ln_b[c0 + j + 3]);
            *(float4*)(dst + j) = v;
        }
    }
}

extern "C" void kernel_launch(void* const* d_in, const int* in_sizes, int n_in,
                              void* d_out, int out_size, void* d_ws, size_t ws_size,
                              hipStream_t stream) {
    (void)n_in; (void)out_size; (void)ws_size;
    const float* x          = (const float*)d_in[0];
    const float* self_Wi    = (const float*)d_in[1];
    const float* self_bi    = (const float*)d_in[2];
    const float* self_Wo    = (const float*)d_in[3];
    const float* self_bo    = (const float*)d_in[4];
    const float* cross_Wi   = (const float*)d_in[5];
    const float* cross_bi   = (const float*)d_in[6];
    const float* cross_Wo   = (const float*)d_in[7];
    const float* cross_bo   = (const float*)d_in[8];
    const float* cross_ln_g = (const float*)d_in[9];
    const float* cross_ln_b = (const float*)d_in[10];
    const float* gate_ln_g  = (const float*)d_in[11];
    const float* gate_ln_b  = (const float*)d_in[12];
    const float* gate_W     = (const float*)d_in[13];
    const float* gate_b     = (const float*)d_in[14];
    const float* final_ln_g = (const float*)d_in[15];
    const float* final_ln_b = (const float*)d_in[16];
    const int*   modal_idx  = (const int*)d_in[17];
    float*       out        = (float*)d_out;

    unsigned short* W_eff = (unsigned short*)d_ws;
    float*          b_eff = (float*)((char*)d_ws + (size_t)NMOD * DD * DD * sizeof(unsigned short));

    fuse_weights_kernel<<<NMOD * (DD / 16), 256, 0, stream>>>(
        self_Wi, self_bi, self_Wo, self_bo,
        cross_Wi, cross_bi, cross_Wo, cross_bo, W_eff, b_eff);

    const int batch = in_sizes[0] / (NMOD * DD);   // 16384
    gcma_main_kernel<<<batch / MT, NTHR, 0, stream>>>(
        x, cross_ln_g, cross_ln_b, gate_ln_g, gate_ln_b, gate_W, gate_b,
        final_ln_g, final_ln_b, W_eff, b_eff, modal_idx, out);
}
